// CHGNet_62895501083196
// MI455X (gfx1250) — compile-verified, matrix operand order checked by reading, not run on hardware
//
#include <hip/hip_runtime.h>

#define D64 64
#define BM 128
#define NTHREADS 256

typedef __attribute__((ext_vector_type(16))) __bf16 bf16x16;
typedef __attribute__((ext_vector_type(8)))  float  f32x8;

__device__ __forceinline__ unsigned short f2bf(float f) {
  union { float f; unsigned u; } x; x.f = f;
  unsigned r = x.u + 0x7FFFu + ((x.u >> 16) & 1u);
  return (unsigned short)(r >> 16);
}
__device__ __forceinline__ float siluf(float x) { return x / (1.f + __expf(-x)); }
__device__ __forceinline__ float sigm(float x)  { return 1.f / (1.f + __expf(-x)); }

__device__ __forceinline__ f32x8 wmma_bf16(bf16x16 a, bf16x16 b, f32x8 c) {
  return __builtin_amdgcn_wmma_f32_16x16x32_bf16(false, a, false, b, (short)0, c, false, false);
}

// A-fragment (16x32 bf16, M x K) from LDS, row-major ushort with leading dim ld.
// ISA layout: lanes 0-15 -> K offset 0, lanes 16-31 -> K offset 8;
// vector elems 0..7 = K kb+0..7, elems 8..15 = K 16+kb+0..7.
// Two contiguous 16B chunks per lane -> 2x ds_load_b128.
__device__ __forceinline__ bf16x16 load_a_frag(const unsigned short* X, int row0, int ld,
                                               int k0, int lane) {
  const unsigned short* p = X + (row0 + (lane & 15)) * ld + k0 + ((lane >> 4) << 3);
  union { bf16x16 v; uint4 q[2]; } r;
  r.q[0] = *(const uint4*)(p);
  r.q[1] = *(const uint4*)(p + 16);
  return r.v;
}

// B-fragment from pre-swizzled weights: fragment f=(kt*4+n) stored as
// [half h][lane l][8 ushorts] -> lane reads two coalesced b128 chunks.
__device__ __forceinline__ bf16x16 load_b_frag(const unsigned short* Wp, int kt, int n, int lane) {
  const uint4* p = (const uint4*)Wp;
  const int f = kt * 4 + n;
  union { bf16x16 v; uint4 q[2]; } r;
  r.q[0] = p[(f * 2 + 0) * 32 + lane];
  r.q[1] = p[(f * 2 + 1) * 32 + lane];
  return r.v;
}

// ---------------------------------------------------------------- utilities
// Repack f32 weight [nmat][K][64] into bf16 WMMA B-fragment order.
// Per matrix, output idx = ((f*2+h)*32 + l)*8 + i, with
//   kt=f>>2, n=f&3, kb=(l>>4)<<3, k=kt*32+h*16+kb+i, col=n*16+(l&15).
__global__ void pack_b_kernel(const float* __restrict__ W, unsigned short* __restrict__ out,
                              int per_mat, int total) {
  int idx = blockIdx.x * NTHREADS + threadIdx.x;
  if (idx >= total) return;
  int m = idx / per_mat;
  int t = idx - m * per_mat;
  int i = t & 7;
  int l = (t >> 3) & 31;
  int h = (t >> 8) & 1;
  int f = t >> 9;
  int kt = f >> 2, n = f & 3;
  int kb = (l >> 4) << 3;
  int k = kt * 32 + h * 16 + kb + i;
  int col = n * 16 + (l & 15);
  out[idx] = f2bf(W[m * per_mat + k * D64 + col]);
}
__global__ void fill0_kernel(float* __restrict__ p, int n) {
  int i = blockIdx.x * NTHREADS + threadIdx.x;
  if (i < n) p[i] = 0.f;
}
__global__ void copy_kernel(float* __restrict__ dst, const float* __restrict__ src, int n) {
  int i = blockIdx.x * NTHREADS + threadIdx.x;
  if (i < n) dst[i] = src[i];
}
__global__ void atom_init_kernel(const float* __restrict__ emb, const int* __restrict__ node_type,
                                 float* __restrict__ atom, int na) {
  int e = blockIdx.x * NTHREADS + threadIdx.x;
  if (e < na * D64) { int a = e >> 6, c = e & 63; atom[e] = emb[node_type[a] * D64 + c]; }
}

// --------------------------------------------------------- featurization
__global__ void __launch_bounds__(NTHREADS)
bond_feat_kernel(const float* __restrict__ bond_vec, const float* __restrict__ freq,
                 const float* __restrict__ embW, const float* __restrict__ embB,
                 const float* __restrict__ abwW,
                 float* __restrict__ bond, float* __restrict__ abw, int nb) {
  __shared__ float sB[NTHREADS][10];
  __shared__ float sEmb[9 * D64];
  __shared__ float sAbw[9 * D64];
  __shared__ float sBias[D64];
  const int tid = threadIdx.x;
  for (int e = tid; e < 9 * D64; e += NTHREADS) { sEmb[e] = embW[e]; sAbw[e] = abwW[e]; }
  if (tid < D64) sBias[tid] = embB[tid];
  const int b = blockIdx.x * NTHREADS + tid;
  if (b < nb) {
    float vx = bond_vec[b * 3 + 0], vy = bond_vec[b * 3 + 1], vz = bond_vec[b * 3 + 2];
    float d = sqrtf(vx * vx + vy * vy + vz * vz);
    float coef = sqrtf(2.f / 5.0f) / d;
#pragma unroll
    for (int n = 0; n < 9; ++n) sB[tid][n] = coef * __sinf(freq[n] * d * (1.f / 5.0f));
  } else {
#pragma unroll
    for (int n = 0; n < 9; ++n) sB[tid][n] = 0.f;
  }
  __syncthreads();
  for (int e = tid; e < NTHREADS * D64; e += NTHREADS) {
    int r = e >> 6, c = e & 63;
    int g = blockIdx.x * NTHREADS + r;
    if (g >= nb) continue;
    float s0 = sBias[c], s1 = 0.f;
#pragma unroll
    for (int n = 0; n < 9; ++n) {
      float t = sB[r][n];
      s0 += t * sEmb[n * D64 + c];
      s1 += t * sAbw[n * D64 + c];
    }
    bond[g * D64 + c] = s0;
    abw[g * D64 + c] = s1;
  }
}

__global__ void __launch_bounds__(NTHREADS)
tbw_feat_kernel(const float* __restrict__ bond_vec, const int* __restrict__ bond_index,
                const float* __restrict__ freq, const float* __restrict__ tbwW,
                float* __restrict__ tbw, int nb3) {
  __shared__ float sB[NTHREADS][10];
  __shared__ float sW[9 * D64];
  const int tid = threadIdx.x;
  for (int e = tid; e < 9 * D64; e += NTHREADS) sW[e] = tbwW[e];
  const int j = blockIdx.x * NTHREADS + tid;
  if (j < nb3) {
    int b = bond_index[j];
    float vx = bond_vec[b * 3 + 0], vy = bond_vec[b * 3 + 1], vz = bond_vec[b * 3 + 2];
    float d = sqrtf(vx * vx + vy * vy + vz * vz);
    float coef = sqrtf(2.f / 3.0f) / d;
#pragma unroll
    for (int n = 0; n < 9; ++n) sB[tid][n] = coef * __sinf(freq[n] * d * (1.f / 3.0f));
  } else {
#pragma unroll
    for (int n = 0; n < 9; ++n) sB[tid][n] = 0.f;
  }
  __syncthreads();
  for (int e = tid; e < NTHREADS * D64; e += NTHREADS) {
    int r = e >> 6, c = e & 63;
    int g = blockIdx.x * NTHREADS + r;
    if (g >= nb3) continue;
    float s = 0.f;
#pragma unroll
    for (int n = 0; n < 9; ++n) s += sB[r][n] * sW[n * D64 + c];
    tbw[g * D64 + c] = s;
  }
}

__global__ void __launch_bounds__(NTHREADS)
angle_feat_kernel(const float* __restrict__ bond_vec, const int* __restrict__ bond_index,
                  const int* __restrict__ angle_src, const int* __restrict__ angle_dst,
                  const float* __restrict__ ffreq, const float* __restrict__ embW,
                  const float* __restrict__ embB, float* __restrict__ angle, int nang) {
  __shared__ float sA[NTHREADS][10];
  __shared__ float sW[9 * D64];
  __shared__ float sBias[D64];
  const int tid = threadIdx.x;
  for (int e = tid; e < 9 * D64; e += NTHREADS) sW[e] = embW[e];
  if (tid < D64) sBias[tid] = embB[tid];
  const int j = blockIdx.x * NTHREADS + tid;
  if (j < nang) {
    int b1 = bond_index[angle_src[j]], b2 = bond_index[angle_dst[j]];
    float ax = bond_vec[b1 * 3 + 0], ay = bond_vec[b1 * 3 + 1], az = bond_vec[b1 * 3 + 2];
    float bx = bond_vec[b2 * 3 + 0], by = bond_vec[b2 * 3 + 1], bz = bond_vec[b2 * 3 + 2];
    float dot = ax * bx + ay * by + az * bz;
    float n1 = sqrtf(ax * ax + ay * ay + az * az);
    float n2 = sqrtf(bx * bx + by * by + bz * bz);
    float cv = dot / (n1 * n2);
    cv = fminf(fmaxf(cv, -1.f + 1e-6f), 1.f - 1e-6f);
    float th = acosf(cv);
    const float isp = 0.5641895835477563f;    // 1/sqrt(pi)
    sA[tid][0] = 0.7071067811865476f * isp;   // (1/sqrt(2))/sqrt(pi)
#pragma unroll
    for (int k = 0; k < 4; ++k) {
      float t = th * ffreq[k];
      sA[tid][1 + k] = __cosf(t) * isp;
      sA[tid][5 + k] = __sinf(t) * isp;
    }
  } else {
#pragma unroll
    for (int n = 0; n < 9; ++n) sA[tid][n] = 0.f;
  }
  __syncthreads();
  for (int e = tid; e < NTHREADS * D64; e += NTHREADS) {
    int r = e >> 6, c = e & 63;
    int g = blockIdx.x * NTHREADS + r;
    if (g >= nang) continue;
    float s = sBias[c];
#pragma unroll
    for (int n = 0; n < 9; ++n) s += sA[r][n] * sW[n * D64 + c];
    angle[g * D64 + c] = s;
  }
}

// ---------------------------------------------- fused atom message block (WMMA)
__global__ void __launch_bounds__(NTHREADS)
atom_block_kernel(const float* __restrict__ atom_in, float* __restrict__ atom_out,
                  const float* __restrict__ bond, const float* __restrict__ abw,
                  const int* __restrict__ src, const int* __restrict__ dstI,
                  const unsigned short* __restrict__ cW1, const float* __restrict__ cb1,
                  const unsigned short* __restrict__ cW2, const float* __restrict__ cb2,
                  const unsigned short* __restrict__ gW1, const float* __restrict__ gb1,
                  const unsigned short* __restrict__ gW2, const float* __restrict__ gb2,
                  int nb) {
  __shared__ __align__(16) unsigned short Xs[BM * 192];
  __shared__ __align__(16) unsigned short Hc[BM * 72];
  __shared__ __align__(16) unsigned short Hg[BM * 72];
  const int tid = threadIdx.x;
  const int base = blockIdx.x * BM;

  // gather e_in = [atom[src], atom[dst], bond] -> LDS bf16
  for (int e = tid; e < BM * 192; e += NTHREADS) {
    int r = e / 192, c = e - r * 192;
    int g = base + r;
    float v = 0.f;
    if (g < nb) {
      if (c < 64)       v = atom_in[src[g] * D64 + c];
      else if (c < 128) v = atom_in[dstI[g] * D64 + (c - 64)];
      else              v = bond[g * D64 + (c - 128)];
    }
    Xs[e] = f2bf(v);
  }
  __syncthreads();

  const int lane = tid & 31;
  const int row0 = (tid >> 5) * 16;
  const int ncol = lane & 15;
  const int mhi  = (lane >> 4) << 3;
  const f32x8 zf = {};

  f32x8 aC[4] = {zf, zf, zf, zf};
  f32x8 aG[4] = {zf, zf, zf, zf};
#pragma unroll
  for (int kt = 0; kt < 6; ++kt) {
    bf16x16 A = load_a_frag(Xs, row0, 192, kt * 32, lane);
#pragma unroll
    for (int n = 0; n < 4; ++n) {
      aC[n] = wmma_bf16(A, load_b_frag(cW1, kt, n, lane), aC[n]);
      aG[n] = wmma_bf16(A, load_b_frag(gW1, kt, n, lane), aG[n]);
    }
  }
#pragma unroll
  for (int n = 0; n < 4; ++n) {
    int col = n * 16 + ncol;
    float bc = cb1[col], bg = gb1[col];
#pragma unroll
    for (int r = 0; r < 8; ++r) {
      int row = row0 + mhi + r;
      Hc[row * 72 + col] = f2bf(siluf(aC[n][r] + bc));
      Hg[row * 72 + col] = f2bf(siluf(aG[n][r] + bg));
    }
  }

  // second layer: each wave reads only the LDS rows it just wrote
  f32x8 oC[4] = {zf, zf, zf, zf};
  f32x8 oG[4] = {zf, zf, zf, zf};
#pragma unroll
  for (int kt = 0; kt < 2; ++kt) {
    bf16x16 Ac = load_a_frag(Hc, row0, 72, kt * 32, lane);
    bf16x16 Ag = load_a_frag(Hg, row0, 72, kt * 32, lane);
#pragma unroll
    for (int n = 0; n < 4; ++n) {
      oC[n] = wmma_bf16(Ac, load_b_frag(cW2, kt, n, lane), oC[n]);
      oG[n] = wmma_bf16(Ag, load_b_frag(gW2, kt, n, lane), oG[n]);
    }
  }
#pragma unroll
  for (int n = 0; n < 4; ++n) {
    int col = n * 16 + ncol;
    float bc = cb2[col], bg = gb2[col];
#pragma unroll
    for (int r = 0; r < 8; ++r) {
      int row = row0 + mhi + r, g = base + row;
      if (g < nb) {
        float m = siluf(oC[n][r] + bc) * sigm(oG[n][r] + bg) * abw[g * D64 + col];
        atomicAdd(&atom_out[dstI[g] * D64 + col], m);   // segment_sum over dst
      }
    }
  }
}

// ------------------------------------------- fused angle/bond message block
__global__ void __launch_bounds__(NTHREADS)
angle_block_kernel(const float* __restrict__ atom, const float* __restrict__ bond,
                   float* __restrict__ angle, float* __restrict__ agg,
                   const int* __restrict__ angle_src, const int* __restrict__ angle_dst,
                   const int* __restrict__ bond_index, const int* __restrict__ dstI,
                   const unsigned short* __restrict__ cW1, const float* __restrict__ cb1,
                   const unsigned short* __restrict__ cW2, const float* __restrict__ cb2,
                   const unsigned short* __restrict__ gW1, const float* __restrict__ gb1,
                   const unsigned short* __restrict__ gW2, const float* __restrict__ gb2,
                   const unsigned short* __restrict__ aW,  const float* __restrict__ ab,
                   const unsigned short* __restrict__ gWa, const float* __restrict__ gba,
                   int nang) {
  __shared__ __align__(16) unsigned short Xs[BM * 256];
  __shared__ __align__(16) unsigned short Hc[BM * 72];
  __shared__ __align__(16) unsigned short Hg[BM * 72];
  __shared__ int rCA[BM], rBS[BM], rBD[BM], rAD[BM];
  const int tid = threadIdx.x;
  const int base = blockIdx.x * BM;

  if (tid < BM) {
    int j = base + tid;
    int as = 0, ad = 0;
    if (j < nang) { as = angle_src[j]; ad = angle_dst[j]; }
    int bs = bond_index[as], bd = bond_index[ad];
    rBS[tid] = bs; rBD[tid] = bd; rCA[tid] = dstI[bd]; rAD[tid] = ad;
  }
  __syncthreads();

  // gather a_in = [atom[center], bond[bidx[as]], bond[bidx[ad]], angle]
  for (int e = tid; e < BM * 256; e += NTHREADS) {
    int r = e >> 8, c = e & 255;
    int g = base + r;
    float v = 0.f;
    if (g < nang) {
      if (c < 64)       v = atom[rCA[r] * D64 + c];
      else if (c < 128) v = bond[rBS[r] * D64 + (c - 64)];
      else if (c < 192) v = bond[rBD[r] * D64 + (c - 128)];
      else              v = angle[g * D64 + (c - 192)];
    }
    Xs[e] = f2bf(v);
  }
  __syncthreads();

  const int lane = tid & 31;
  const int row0 = (tid >> 5) * 16;
  const int ncol = lane & 15;
  const int mhi  = (lane >> 4) << 3;
  const f32x8 zf = {};

  { // pass 1: bond-message hidden layer -> LDS
    f32x8 aC[4] = {zf, zf, zf, zf};
    f32x8 aG[4] = {zf, zf, zf, zf};
#pragma unroll
    for (int kt = 0; kt < 8; ++kt) {
      bf16x16 A = load_a_frag(Xs, row0, 256, kt * 32, lane);
#pragma unroll
      for (int n = 0; n < 4; ++n) {
        aC[n] = wmma_bf16(A, load_b_frag(cW1, kt, n, lane), aC[n]);
        aG[n] = wmma_bf16(A, load_b_frag(gW1, kt, n, lane), aG[n]);
      }
    }
#pragma unroll
    for (int n = 0; n < 4; ++n) {
      int col = n * 16 + ncol;
      float bc = cb1[col], bg = gb1[col];
#pragma unroll
      for (int r = 0; r < 8; ++r) {
        int row = row0 + mhi + r;
        Hc[row * 72 + col] = f2bf(siluf(aC[n][r] + bc));
        Hg[row * 72 + col] = f2bf(siluf(aG[n][r] + bg));
      }
    }
  }

  { // pass 2: angle self-update (single gated layer), row-owned in-place add
    f32x8 pC[4] = {zf, zf, zf, zf};
    f32x8 pG[4] = {zf, zf, zf, zf};
#pragma unroll
    for (int kt = 0; kt < 8; ++kt) {
      bf16x16 A = load_a_frag(Xs, row0, 256, kt * 32, lane);
#pragma unroll
      for (int n = 0; n < 4; ++n) {
        pC[n] = wmma_bf16(A, load_b_frag(aW,  kt, n, lane), pC[n]);
        pG[n] = wmma_bf16(A, load_b_frag(gWa, kt, n, lane), pG[n]);
      }
    }
#pragma unroll
    for (int n = 0; n < 4; ++n) {
      int col = n * 16 + ncol;
      float bc = ab[col], bg = gba[col];
#pragma unroll
      for (int r = 0; r < 8; ++r) {
        int row = row0 + mhi + r, g = base + row;
        if (g < nang)
          angle[g * D64 + col] += siluf(pC[n][r] + bc) * sigm(pG[n][r] + bg);
      }
    }
  }

  { // pass 3: bond-message output layer + segment_sum by angle_dst into agg
    f32x8 oC[4] = {zf, zf, zf, zf};
    f32x8 oG[4] = {zf, zf, zf, zf};
#pragma unroll
    for (int kt = 0; kt < 2; ++kt) {
      bf16x16 Ac = load_a_frag(Hc, row0, 72, kt * 32, lane);
      bf16x16 Ag = load_a_frag(Hg, row0, 72, kt * 32, lane);
#pragma unroll
      for (int n = 0; n < 4; ++n) {
        oC[n] = wmma_bf16(Ac, load_b_frag(cW2, kt, n, lane), oC[n]);
        oG[n] = wmma_bf16(Ag, load_b_frag(gW2, kt, n, lane), oG[n]);
      }
    }
#pragma unroll
    for (int n = 0; n < 4; ++n) {
      int col = n * 16 + ncol;
      float bc = cb2[col], bg = gb2[col];
#pragma unroll
      for (int r = 0; r < 8; ++r) {
        int row = row0 + mhi + r, g = base + row;
        if (g < nang) {
          float m = siluf(oC[n][r] + bc) * sigm(oG[n][r] + bg);
          atomicAdd(&agg[rAD[row] * D64 + col], m);
        }
      }
    }
  }
}

__global__ void bond_apply_kernel(float* __restrict__ bond, const float* __restrict__ agg,
                                  const float* __restrict__ tbw,
                                  const int* __restrict__ bond_index, int n) {
  int e = blockIdx.x * NTHREADS + threadIdx.x;
  if (e >= n) return;
  int j = e >> 6, c = e & 63;
  atomicAdd(&bond[bond_index[j] * D64 + c], agg[e] * tbw[e]);
}

// ----------------------------------------------------------------- readout
__global__ void __launch_bounds__(NTHREADS)
readout_kernel(const float* __restrict__ atom, const float* __restrict__ siteW,
               const float* __restrict__ siteB, const float* __restrict__ W1,
               const float* __restrict__ b1, const float* __restrict__ W2,
               const float* __restrict__ b2, const float* __restrict__ W3,
               const float* __restrict__ b3, float* __restrict__ out, int na) {
  __shared__ float sW1[64 * 64];
  __shared__ float sW2[64 * 64];
  __shared__ float sX[4][64];
  __shared__ float sH[4][64];
  __shared__ float sR[4][64];
  const int tid = threadIdx.x;
  for (int e = tid; e < 64 * 64; e += NTHREADS) { sW1[e] = W1[e]; sW2[e] = W2[e]; }
  const int c = tid & 63, q = tid >> 6;
  const int a = blockIdx.x * 4 + q;
  float x = (a < na) ? atom[a * D64 + c] : 0.f;
  sX[q][c] = x;
  __syncthreads();
  float s = b1[c];
#pragma unroll 8
  for (int k = 0; k < 64; ++k) s += sX[q][k] * sW1[k * 64 + c];
  sH[q][c] = siluf(s);
  __syncthreads();
  float s2 = b2[c];
#pragma unroll 8
  for (int k = 0; k < 64; ++k) s2 += sH[q][k] * sW2[k * 64 + c];
  float h2 = siluf(s2);
  // site = atom @ site_W + site_b
  sR[q][c] = x * siteW[c];
  __syncthreads();
  for (int o = 32; o >= 1; o >>= 1) {
    if (c < o) sR[q][c] += sR[q][c + o];
    __syncthreads();
  }
  if (c == 0 && a < na) out[1 + a] = sR[q][0] + siteB[0];
  __syncthreads();
  sR[q][c] = h2 * W3[c];
  __syncthreads();
  for (int o = 32; o >= 1; o >>= 1) {
    if (c < o) sR[q][c] += sR[q][c + o];
    __syncthreads();
  }
  if (c == 0 && a < na) atomicAdd(out, sR[q][0] + b3[0]);
}

// ------------------------------------------------------------------- host
extern "C" void kernel_launch(void* const* d_in, const int* in_sizes, int n_in,
                              void* d_out, int out_size, void* d_ws, size_t ws_size,
                              hipStream_t stream) {
  // inputs in setup_inputs() dict insertion order (params dict first, recursively)
  const float* freq_rbf     = (const float*)d_in[0];
  const float* freq_rbf3    = (const float*)d_in[1];
  const float* freq_fourier = (const float*)d_in[2];
  const float* atom_emb     = (const float*)d_in[3];
  const float* bond_emb_W   = (const float*)d_in[4];
  const float* bond_emb_b   = (const float*)d_in[5];
  const float* angle_emb_W  = (const float*)d_in[6];
  const float* angle_emb_b  = (const float*)d_in[7];
  const float* abw_W        = (const float*)d_in[8];
  const float* tbw_W        = (const float*)d_in[9];
  const float* atom_cW1 = (const float*)d_in[10];
  const float* atom_cb1 = (const float*)d_in[11];
  const float* atom_cW2 = (const float*)d_in[12];
  const float* atom_cb2 = (const float*)d_in[13];
  const float* atom_gW1 = (const float*)d_in[14];
  const float* atom_gb1 = (const float*)d_in[15];
  const float* atom_gW2 = (const float*)d_in[16];
  const float* atom_gb2 = (const float*)d_in[17];
  const float* bond_cW1 = (const float*)d_in[18];
  const float* bond_cb1 = (const float*)d_in[19];
  const float* bond_cW2 = (const float*)d_in[20];
  const float* bond_cb2 = (const float*)d_in[21];
  const float* bond_gW1 = (const float*)d_in[22];
  const float* bond_gb1 = (const float*)d_in[23];
  const float* bond_gW2 = (const float*)d_in[24];
  const float* bond_gb2 = (const float*)d_in[25];
  const float* ang_cW   = (const float*)d_in[26];
  const float* ang_cb   = (const float*)d_in[27];
  const float* ang_gW   = (const float*)d_in[28];
  const float* ang_gb   = (const float*)d_in[29];
  const float* site_W   = (const float*)d_in[30];
  const float* site_b   = (const float*)d_in[31];
  const float* ro_W1    = (const float*)d_in[32];
  const float* ro_b1    = (const float*)d_in[33];
  const float* ro_W2    = (const float*)d_in[34];
  const float* ro_b2    = (const float*)d_in[35];
  const float* ro_W3    = (const float*)d_in[36];
  const float* ro_b3    = (const float*)d_in[37];
  const float* bond_vec = (const float*)d_in[38];
  const int* node_type  = (const int*)d_in[39];
  const int* src        = (const int*)d_in[40];
  const int* dstI       = (const int*)d_in[41];
  const int* bond_index = (const int*)d_in[42];
  const int* angle_src  = (const int*)d_in[43];
  const int* angle_dst  = (const int*)d_in[44];
  (void)n_in; (void)out_size; (void)ws_size;

  const int na   = in_sizes[39];
  const int nb   = in_sizes[40];
  const int nb3  = in_sizes[42];
  const int nang = in_sizes[43];

  char* ws = (char*)d_ws;
  size_t off = 0;
  auto take = [&](size_t bytes) -> char* {
    char* p = ws + off;
    off = (off + bytes + 255) & ~(size_t)255;
    return p;
  };
  unsigned short* wAc1 = (unsigned short*)take((size_t)4 * 192 * 64 * 2);
  unsigned short* wAg1 = (unsigned short*)take((size_t)4 * 192 * 64 * 2);
  unsigned short* wAc2 = (unsigned short*)take((size_t)4 * 64 * 64 * 2);
  unsigned short* wAg2 = (unsigned short*)take((size_t)4 * 64 * 64 * 2);
  unsigned short* wBc1 = (unsigned short*)take((size_t)3 * 256 * 64 * 2);
  unsigned short* wBg1 = (unsigned short*)take((size_t)3 * 256 * 64 * 2);
  unsigned short* wBc2 = (unsigned short*)take((size_t)3 * 64 * 64 * 2);
  unsigned short* wBg2 = (unsigned short*)take((size_t)3 * 64 * 64 * 2);
  unsigned short* wNc  = (unsigned short*)take((size_t)3 * 256 * 64 * 2);
  unsigned short* wNg  = (unsigned short*)take((size_t)3 * 256 * 64 * 2);
  float* atomA = (float*)take((size_t)na * 64 * 4);
  float* atomB = (float*)take((size_t)na * 64 * 4);
  float* bond  = (float*)take((size_t)nb * 64 * 4);
  float* abw   = (float*)take((size_t)nb * 64 * 4);
  float* angle = (float*)take((size_t)nang * 64 * 4);
  float* tbw   = (float*)take((size_t)nb3 * 64 * 4);
  float* agg   = (float*)take((size_t)nb3 * 64 * 4);

  // repack weights into WMMA B-fragment order (bf16)
  auto pack = [&](const float* s, unsigned short* d, int K, int nmat) {
    int total = nmat * K * D64;
    pack_b_kernel<<<(total + NTHREADS - 1) / NTHREADS, NTHREADS, 0, stream>>>(s, d, K * D64,
                                                                              total);
  };
  pack(atom_cW1, wAc1, 192, 4);
  pack(atom_gW1, wAg1, 192, 4);
  pack(atom_cW2, wAc2, 64, 4);
  pack(atom_gW2, wAg2, 64, 4);
  pack(bond_cW1, wBc1, 256, 3);
  pack(bond_gW1, wBg1, 256, 3);
  pack(bond_cW2, wBc2, 64, 3);
  pack(bond_gW2, wBg2, 64, 3);
  pack(ang_cW,   wNc,  256, 3);
  pack(ang_gW,   wNg,  256, 3);

  bond_feat_kernel<<<(nb + NTHREADS - 1) / NTHREADS, NTHREADS, 0, stream>>>(
      bond_vec, freq_rbf, bond_emb_W, bond_emb_b, abw_W, bond, abw, nb);
  tbw_feat_kernel<<<(nb3 + NTHREADS - 1) / NTHREADS, NTHREADS, 0, stream>>>(
      bond_vec, bond_index, freq_rbf3, tbw_W, tbw, nb3);
  angle_feat_kernel<<<(nang + NTHREADS - 1) / NTHREADS, NTHREADS, 0, stream>>>(
      bond_vec, bond_index, angle_src, angle_dst, freq_fourier, angle_emb_W, angle_emb_b,
      angle, nang);
  atom_init_kernel<<<(na * 64 + NTHREADS - 1) / NTHREADS, NTHREADS, 0, stream>>>(
      atom_emb, node_type, atomA, na);

  float* cur = atomA;
  float* nxt = atomB;
  for (int i = 0; i < 4; ++i) {
    copy_kernel<<<(na * 64 + NTHREADS - 1) / NTHREADS, NTHREADS, 0, stream>>>(nxt, cur, na * 64);
    atom_block_kernel<<<(nb + BM - 1) / BM, NTHREADS, 0, stream>>>(
        cur, nxt, bond, abw, src, dstI,
        wAc1 + (size_t)i * 192 * 64, atom_cb1 + i * 64,
        wAc2 + (size_t)i * 64 * 64,  atom_cb2 + i * 64,
        wAg1 + (size_t)i * 192 * 64, atom_gb1 + i * 64,
        wAg2 + (size_t)i * 64 * 64,  atom_gb2 + i * 64, nb);
    float* t = cur; cur = nxt; nxt = t;
    if (i < 3) {
      fill0_kernel<<<(nb3 * 64 + NTHREADS - 1) / NTHREADS, NTHREADS, 0, stream>>>(agg, nb3 * 64);
      angle_block_kernel<<<(nang + BM - 1) / BM, NTHREADS, 0, stream>>>(
          cur, bond, angle, agg, angle_src, angle_dst, bond_index, dstI,
          wBc1 + (size_t)i * 256 * 64, bond_cb1 + i * 64,
          wBc2 + (size_t)i * 64 * 64,  bond_cb2 + i * 64,
          wBg1 + (size_t)i * 256 * 64, bond_gb1 + i * 64,
          wBg2 + (size_t)i * 64 * 64,  bond_gb2 + i * 64,
          wNc  + (size_t)i * 256 * 64, ang_cb + i * 64,
          wNg  + (size_t)i * 256 * 64, ang_gb + i * 64, nang);
      bond_apply_kernel<<<(nb3 * 64 + NTHREADS - 1) / NTHREADS, NTHREADS, 0, stream>>>(
          bond, agg, tbw, bond_index, nb3 * 64);
    }
  }

  fill0_kernel<<<1, NTHREADS, 0, stream>>>((float*)d_out, 1);
  readout_kernel<<<(na + 3) / 4, NTHREADS, 0, stream>>>(
      cur, site_W, site_b, ro_W1, ro_b1, ro_W2, ro_b2, ro_W3, ro_b3, (float*)d_out, na);
}
